// RolandGNN_44117904065162
// MI455X (gfx1250) — matrix operand order repaired and checked by reading, not compiled, not run on hardware
//
#include <hip/hip_runtime.h>

#define TDIN 128
#define TH1  256
#define TH2  128
#define LEAK 0.01f

typedef __bf16        v16bf __attribute__((ext_vector_type(16)));
typedef float         v8f   __attribute__((ext_vector_type(8)));
typedef unsigned int  u32x4 __attribute__((ext_vector_type(4)));

union FragBF { v16bf v; u32x4 q[2]; };

__device__ __forceinline__ unsigned short f2b(float x) {
  unsigned int u = __float_as_uint(x);
  unsigned int r = u + 0x7FFFu + ((u >> 16) & 1u);   // round-to-nearest-even
  return (unsigned short)(r >> 16);
}

// ---- CDNA5 async global->LDS copy (16B per lane), tracked by ASYNCcnt ----
__device__ __forceinline__ void async_cp16(const void* g, unsigned int loff) {
  asm volatile("global_load_async_to_lds_b128 %0, %1, off"
               :: "v"(loff), "v"(g) : "memory");
}
__device__ __forceinline__ void async_wait_le1() {
  asm volatile("s_wait_asynccnt 0x1" ::: "memory");
}
__device__ __forceinline__ void async_wait_le0() {
  asm volatile("s_wait_asynccnt 0x0" ::: "memory");
}

// ---------------------------------------------------------------------------
// WMMA GEMM:  out[N,M] = act(A[N,K] @ W[K,M] + bias)
// A row-major bf16; Wp pre-packed B fragments (k_pack_B). Compile-time K/M/NT.
// Block: 8 waves, tile 64 x (32*NT); K fully unrolled in 32-chunks;
// triple-buffered async A staging (one barrier per chunk); NT acc / wave.
// ---------------------------------------------------------------------------
template<int K, int M, int NT>
__global__ __launch_bounds__(256) void k_gemm_bf16(
    const unsigned short* __restrict__ A,
    const unsigned short* __restrict__ Wp,
    const float* __restrict__ bias,
    float* __restrict__ outF,
    unsigned short* __restrict__ outB,
    int Nrows, int act)
{
  constexpr int KC = K / 32;
  __shared__ unsigned short As[3][64 * 32];   // 3 x 4KB rotating

  const int tid  = threadIdx.x;
  const int lane = tid & 31;
  const int wv   = tid >> 5;
  const int rt   = wv & 3;              // wave's 16-row tile
  const int cg   = wv >> 2;             // wave's column group (0/1)
  const int row0 = blockIdx.x * 64;
  const int ct0  = blockIdx.y * (2 * NT);
  const int mrow = lane & 15;
  const int hf   = lane >> 4;

  // A staging role: 64 rows x 32 k, 16B per thread; clamp OOB rows (never written back)
  const int ar   = tid >> 2;
  const int aseg = (tid & 3) * 8;
  int gr = row0 + ar; if (gr >= Nrows) gr = Nrows - 1;
  const unsigned short* aG = A + (size_t)gr * K + aseg;
  const unsigned int aL[3] = {
    (unsigned int)(size_t)&As[0][ar * 32 + aseg],
    (unsigned int)(size_t)&As[1][ar * 32 + aseg],
    (unsigned int)(size_t)&As[2][ar * 32 + aseg] };

  // per-tile packed-B base pointers (advance by immediate kc*512 elems)
  const unsigned short* bbase[NT];
#pragma unroll
  for (int t = 0; t < NT; ++t)
    bbase[t] = Wp + ((size_t)((ct0 + cg + 2 * t) * KC) * 32 + lane) * 16;

  v8f acc[NT];
#pragma unroll
  for (int t = 0; t < NT; ++t) acc[t] = v8f{0.f,0.f,0.f,0.f,0.f,0.f,0.f,0.f};

  async_cp16(aG, aL[0]);                 // prefetch chunks 0 and 1
  async_cp16(aG + 32, aL[1]);

#pragma unroll
  for (int kc = 0; kc < KC; ++kc) {
    // async loads complete in order: <=1 outstanding ==> stage(kc) landed
    if (kc + 2 <= KC) async_wait_le1(); else async_wait_le0();
    __syncthreads();
    if (kc + 2 < KC) async_cp16(aG + (kc + 2) * 32, aL[(kc + 2) % 3]);

    // A fragment (16x32 bf16): K = {16h..16h+15}
    FragBF a;
    const unsigned short* ap = &As[kc % 3][(rt * 16 + mrow) * 32 + hf * 16];
    a.q[0] = *(const u32x4*)(ap);
    a.q[1] = *(const u32x4*)(ap + 8);

#pragma unroll
    for (int t = 0; t < NT; ++t) {
      const unsigned short* bp = bbase[t] + kc * 512;   // 32 lanes * 16 elems
      FragBF b;
      b.q[0] = *(const u32x4*)(bp);
      b.q[1] = *(const u32x4*)(bp + 8);
      acc[t] = __builtin_amdgcn_wmma_f32_16x16x32_bf16(false, a.v, false, b.v,
                                                       (short)0, acc[t], false, false);
    }
  }

  // epilogue: bias + leaky, write f32 and/or bf16
#pragma unroll
  for (int t = 0; t < NT; ++t) {
    const int col = (ct0 + cg + 2 * t) * 16 + mrow;
    const float bv = bias ? bias[col] : 0.0f;
#pragma unroll
    for (int r = 0; r < 8; ++r) {
      const int row = row0 + rt * 16 + hf * 8 + r;
      if (row < Nrows) {
        float v = acc[t][r] + bv;
        if (act) v = v > 0.0f ? v : v * LEAK;
        const size_t o = (size_t)row * M + col;
        if (outF) outF[o] = v;
        if (outB) outB[o] = f2b(v);
      }
    }
  }
}

// ---------------------------------------------------------------------------
// Pack weights (f32 -> bf16) into B-fragment layout:
//   D[((ct*KC + kc)*32 + lane)*16 + e] = W[k, m],  k = kc*32 + (lane>>4)*16 + e,
//   m = ct*16 + (lane&15).  trans=1: source is [M,K] row-major (take W^T).
// ---------------------------------------------------------------------------
__global__ void k_pack_B(const float* __restrict__ S, unsigned short* __restrict__ D,
                         int K, int M, int trans) {
  size_t i = (size_t)blockIdx.x * 256 + threadIdx.x;
  if (i >= (size_t)K * M) return;
  int e    = (int)(i & 15);
  int lane = (int)((i >> 4) & 31);
  int tk   = (int)(i >> 9);
  int KC   = K >> 5;
  int kc   = tk % KC;
  int ct   = tk / KC;
  int k = kc * 32 + (lane >> 4) * 16 + e;
  int m = ct * 16 + (lane & 15);
  float v = trans ? S[(size_t)m * K + k] : S[(size_t)k * M + m];
  D[i] = f2b(v);
}

// ---------------------------------------------------------------------------
// Elementwise / graph kernels
// ---------------------------------------------------------------------------
__global__ void k_cvt_bf16(const float* __restrict__ s, unsigned short* __restrict__ d, size_t n) {
  size_t i = (size_t)blockIdx.x * 256 + threadIdx.x;
  if (i < n) d[i] = f2b(s[i]);
}

__global__ void k_zero_f32(float* __restrict__ p, size_t n) {
  size_t i = (size_t)blockIdx.x * 256 + threadIdx.x;
  if (i < n) p[i] = 0.0f;
}
__global__ void k_zero_u32(unsigned int* __restrict__ p, size_t n) {
  size_t i = (size_t)blockIdx.x * 256 + threadIdx.x;
  if (i < n) p[i] = 0u;
}

__global__ void k_deg(const int* __restrict__ dst, unsigned int* __restrict__ deg, int E) {
  size_t i = (size_t)blockIdx.x * 256 + threadIdx.x;
  if (i < (size_t)E) atomicAdd(&deg[dst[i]], 1u);
}
__global__ void k_dis(const unsigned int* __restrict__ deg, float* __restrict__ dis, int N) {
  size_t i = (size_t)blockIdx.x * 256 + threadIdx.x;
  if (i < (size_t)N) dis[i] = rsqrtf((float)deg[i] + 1.0f);  // +1 = self loop
}

// per-(edge, 4-channel chunk): agg[dst] += hw[src] * dis[src]*dis[dst]
__global__ void k_scatter(const int* __restrict__ src, const int* __restrict__ dst,
                          const float* __restrict__ dis, const float* __restrict__ hw,
                          float* __restrict__ agg, int E, int M, int shift /*log2(M/4)*/) {
  size_t idx = (size_t)blockIdx.x * 256 + threadIdx.x;
  int e = (int)(idx >> shift);
  if (e >= E) return;
  int c = (int)(idx & (((size_t)1 << shift) - 1)) * 4;
  int s = src[e], d = dst[e];
  float coef = dis[s] * dis[d];
  const float4 hv = *(const float4*)(hw + (size_t)s * M + c);
  float* ap = agg + (size_t)d * M + c;
  unsafeAtomicAdd(ap + 0, hv.x * coef);
  unsafeAtomicAdd(ap + 1, hv.y * coef);
  unsafeAtomicAdd(ap + 2, hv.z * coef);
  unsafeAtomicAdd(ap + 3, hv.w * coef);
}

// GCN epilogue: v = agg + hw*dis^2 (self loop) + bias; leaky; emit bf16
__global__ void k_gcn_finish(const float* __restrict__ agg, const float* __restrict__ hw,
                             const float* __restrict__ dis, const float* __restrict__ bias,
                             unsigned short* __restrict__ outB, int N, int M, int shift) {
  size_t i = (size_t)blockIdx.x * 256 + threadIdx.x;
  int n = (int)(i >> shift);
  if (n >= N) return;
  int c = (int)(i & (M - 1));
  float dd = dis[n];
  float v = agg[i] + hw[i] * dd * dd + bias[c];
  v = v > 0.0f ? v : v * LEAK;
  outB[i] = f2b(v);
}

__device__ __forceinline__ float sigm(float x) { return 1.0f / (1.0f + __expf(-x)); }

__global__ void k_gru(const float* __restrict__ gi, const float* __restrict__ gh,
                      const float* __restrict__ prev, float* __restrict__ outF,
                      unsigned short* __restrict__ outB, int N, int H, int shift) {
  size_t i = (size_t)blockIdx.x * 256 + threadIdx.x;
  int n = (int)(i >> shift);
  if (n >= N) return;
  int c = (int)(i & (H - 1));
  size_t b = (size_t)n * 3 * H + c;
  float ir = gi[b], iz = gi[b + H], in = gi[b + 2 * H];
  float hr = gh[b], hz = gh[b + H], hn = gh[b + 2 * H];
  float hp = prev[i];
  float r = sigm(ir + hr);
  float z = sigm(iz + hz);
  float nn = tanhf(in + r * hn);
  float h = (1.0f - z) * nn + z * hp;
  outF[i] = h;
  if (outB) outB[i] = f2b(h);
}

// out[n] = sum_c emb1[n,c]*(Wpost[c,0]+Wpost[c,1]) + (bpost0+bpost1); wave per node
__global__ void k_post(const float* __restrict__ emb1, const float* __restrict__ Wpost,
                       const float* __restrict__ bpost, float* __restrict__ out, int N) {
  int wv = threadIdx.x >> 5, lane = threadIdx.x & 31;
  int node = blockIdx.x * 8 + wv;
  if (node >= N) return;
  float s = 0.0f;
  for (int c = lane; c < TH2; c += 32)
    s += emb1[(size_t)node * TH2 + c] * (Wpost[c * 2] + Wpost[c * 2 + 1]);
  for (int off = 16; off > 0; off >>= 1) s += __shfl_down(s, off, 32);
  if (lane == 0) out[node] = s + bpost[0] + bpost[1];
}

// ---------------------------------------------------------------------------
extern "C" void kernel_launch(void* const* d_in, const int* in_sizes, int n_in,
                              void* d_out, int out_size, void* d_ws, size_t ws_size,
                              hipStream_t stream) {
  (void)n_in; (void)out_size; (void)ws_size;
  const float* x     = (const float*)d_in[0];
  const int*   ei    = (const int*)d_in[1];
  const float* prev0 = (const float*)d_in[2];
  const float* prev1 = (const float*)d_in[3];
  const float* Wp1 = (const float*)d_in[4];  const float* bp1 = (const float*)d_in[5];
  const float* Wp2 = (const float*)d_in[6];  const float* bp2 = (const float*)d_in[7];
  const float* Wc1 = (const float*)d_in[8];  const float* bc1 = (const float*)d_in[9];
  const float* Wc2 = (const float*)d_in[10]; const float* bc2 = (const float*)d_in[11];
  const float* Wih1 = (const float*)d_in[12]; const float* Whh1 = (const float*)d_in[13];
  const float* bih1 = (const float*)d_in[14]; const float* bhh1 = (const float*)d_in[15];
  const float* Wih2 = (const float*)d_in[16]; const float* Whh2 = (const float*)d_in[17];
  const float* bih2 = (const float*)d_in[18]; const float* bhh2 = (const float*)d_in[19];
  const float* Wpost = (const float*)d_in[20]; const float* bpost = (const float*)d_in[21];

  const int N = in_sizes[0] / TDIN;
  const int E = in_sizes[1] / 2;
  const int* srcI = ei;
  const int* dstI = ei + E;

  // ---- workspace bump allocator (256B aligned) ----
  char* ws = (char*)d_ws;
  size_t off = 0;
  auto alloc = [&](size_t bytes) -> void* {
    void* p = ws + off;
    off += (bytes + 255) & ~(size_t)255;
    return p;
  };
  unsigned short* xb   = (unsigned short*)alloc((size_t)N * TDIN * 2);
  unsigned short* p0b  = (unsigned short*)alloc((size_t)N * TH1 * 2);
  unsigned short* p1b  = (unsigned short*)alloc((size_t)N * TH2 * 2);
  unsigned short* t0   = (unsigned short*)alloc((size_t)N * TH1 * 2);  // rotating bf16
  unsigned short* t1   = (unsigned short*)alloc((size_t)N * TH1 * 2);
  float* f0 = (float*)alloc((size_t)N * TH1 * 4);                      // hw
  float* f1 = (float*)alloc((size_t)N * TH1 * 4);                      // agg
  float* g0 = (float*)alloc((size_t)N * 3 * TH1 * 4);                  // gi
  float* g1 = (float*)alloc((size_t)N * 3 * TH1 * 4);                  // gh
  unsigned int* deg = (unsigned int*)alloc((size_t)N * 4);
  float* dis = (float*)alloc((size_t)N * 4);
  unsigned short* wp1b  = (unsigned short*)alloc((size_t)TDIN * 256 * 2);
  unsigned short* wp2b  = (unsigned short*)alloc((size_t)256 * TDIN * 2);
  unsigned short* wc1b  = (unsigned short*)alloc((size_t)TDIN * TH1 * 2);
  unsigned short* wc2b  = (unsigned short*)alloc((size_t)TH1 * TH2 * 2);
  unsigned short* wih1t = (unsigned short*)alloc((size_t)TH1 * 3 * TH1 * 2);
  unsigned short* whh1t = (unsigned short*)alloc((size_t)TH1 * 3 * TH1 * 2);
  unsigned short* wih2t = (unsigned short*)alloc((size_t)TH2 * 3 * TH2 * 2);
  unsigned short* whh2t = (unsigned short*)alloc((size_t)TH2 * 3 * TH2 * 2);

  float* outv = (float*)d_out;                    // [N]
  float* emb0 = outv + N;                         // [N,H1]
  float* emb1 = emb0 + (size_t)N * TH1;           // [N,H2]

  auto blks = [](size_t n) { return (unsigned)((n + 255) / 256); };

#define GEMM(Kc, Mc, NTc, A_, W_, b_, oF_, oB_, act_)                                   \
  k_gemm_bf16<Kc, Mc, NTc><<<dim3((unsigned)((N + 63) / 64), (unsigned)(Mc / (32 * NTc))), \
                             256, 0, stream>>>(A_, W_, b_, oF_, oB_, N, act_)

  // ---- activation conversions ----
  k_cvt_bf16<<<blks((size_t)N * TDIN), 256, 0, stream>>>(x, xb, (size_t)N * TDIN);
  k_cvt_bf16<<<blks((size_t)N * TH1), 256, 0, stream>>>(prev0, p0b, (size_t)N * TH1);
  k_cvt_bf16<<<blks((size_t)N * TH2), 256, 0, stream>>>(prev1, p1b, (size_t)N * TH2);

  // ---- weight packing into B-fragment layout ----
  k_pack_B<<<blks((size_t)TDIN * 256), 256, 0, stream>>>(Wp1, wp1b, TDIN, 256, 0);
  k_pack_B<<<blks((size_t)256 * TDIN), 256, 0, stream>>>(Wp2, wp2b, 256, TDIN, 0);
  k_pack_B<<<blks((size_t)TDIN * TH1), 256, 0, stream>>>(Wc1, wc1b, TDIN, TH1, 0);
  k_pack_B<<<blks((size_t)TH1 * TH2), 256, 0, stream>>>(Wc2, wc2b, TH1, TH2, 0);
  k_pack_B<<<blks((size_t)TH1 * 3 * TH1), 256, 0, stream>>>(Wih1, wih1t, TH1, 3 * TH1, 1);
  k_pack_B<<<blks((size_t)TH1 * 3 * TH1), 256, 0, stream>>>(Whh1, whh1t, TH1, 3 * TH1, 1);
  k_pack_B<<<blks((size_t)TH2 * 3 * TH2), 256, 0, stream>>>(Wih2, wih2t, TH2, 3 * TH2, 1);
  k_pack_B<<<blks((size_t)TH2 * 3 * TH2), 256, 0, stream>>>(Whh2, whh2t, TH2, 3 * TH2, 1);

  // ---- degree / normalization ----
  k_zero_u32<<<blks(N), 256, 0, stream>>>(deg, (size_t)N);
  k_deg<<<blks(E), 256, 0, stream>>>(dstI, deg, E);
  k_dis<<<blks(N), 256, 0, stream>>>(deg, dis, N);

  // ---- preprocess MLP ----
  GEMM(TDIN, 256, 8, xb, wp1b, bp1, nullptr, t0, 1);     // h1 = leaky(x@Wp1+b)
  GEMM(256, TDIN, 4, t0, wp2b, bp2, nullptr, t1, 1);     // h2 = leaky(h1@Wp2+b)

  // ---- GCN layer 1 ----
  GEMM(TDIN, TH1, 8, t1, wc1b, nullptr, f0, nullptr, 0); // hw1
  k_zero_f32<<<blks((size_t)N * TH1), 256, 0, stream>>>(f1, (size_t)N * TH1);
  k_scatter<<<blks((size_t)E * (TH1 / 4)), 256, 0, stream>>>(srcI, dstI, dis, f0, f1, E, TH1, 6);
  k_gcn_finish<<<blks((size_t)N * TH1), 256, 0, stream>>>(f1, f0, dis, bc1, t0, N, TH1, 8);

  // ---- GRU 1 ----
  GEMM(TH1, 3 * TH1, 8, t0,  wih1t, bih1, g0, nullptr, 0);  // gi1
  GEMM(TH1, 3 * TH1, 8, p0b, whh1t, bhh1, g1, nullptr, 0);  // gh1
  k_gru<<<blks((size_t)N * TH1), 256, 0, stream>>>(g0, g1, prev0, emb0, t1, N, TH1, 8);

  // ---- GCN layer 2 ----
  GEMM(TH1, TH2, 4, t1, wc2b, nullptr, f0, nullptr, 0);  // hw2
  k_zero_f32<<<blks((size_t)N * TH2), 256, 0, stream>>>(f1, (size_t)N * TH2);
  k_scatter<<<blks((size_t)E * (TH2 / 4)), 256, 0, stream>>>(srcI, dstI, dis, f0, f1, E, TH2, 5);
  k_gcn_finish<<<blks((size_t)N * TH2), 256, 0, stream>>>(f1, f0, dis, bc2, t0, N, TH2, 7);

  // ---- GRU 2 ----
  GEMM(TH2, 3 * TH2, 4, t0,  wih2t, bih2, g0, nullptr, 0);  // gi2
  GEMM(TH2, 3 * TH2, 4, p1b, whh2t, bhh2, g1, nullptr, 0);  // gh2
  k_gru<<<blks((size_t)N * TH2), 256, 0, stream>>>(g0, g1, prev1, emb1, nullptr, N, TH2, 7);

  // ---- postprocess ----
  k_post<<<(unsigned)((N + 7) / 8), 256, 0, stream>>>(emb1, Wpost, bpost, outv, N);
#undef GEMM
}
